// AbsorbanceLookup_49632642072856
// MI455X (gfx1250) — matrix operand order repaired and checked by reading, not compiled, not run on hardware
//
#include <hip/hip_runtime.h>
#include <math.h>

typedef __attribute__((ext_vector_type(16))) _Float16 v16h;
typedef __attribute__((ext_vector_type(8)))  float    v8f;

#define N_CONCS 16
#define N_WL    601
#define WAVES   8   // waves per block (wave32)

// Branch-free tanh: tanh(x) = sign(x) * (1 - e) / (1 + e), e = exp(-2|x|).
// Argument of exp is <= 0: never overflows; underflow -> e = 0 -> tanh -> +/-1.
// v_exp_f32 + v_rcp_f32, no exec-mask divergence (unlike libm tanhf).
__device__ __forceinline__ float fast_tanh(float x) {
    float ax = fabsf(x);
    float e  = __expf(-2.0f * ax);
    float t  = (1.0f - e) * __builtin_amdgcn_rcpf(1.0f + e);
    return copysignf(t, x);
}

// ---------------------------------------------------------------------------
// Kernel 1: mean / std of abs_matrix (16*601 = 9616 floats), population std.
// ---------------------------------------------------------------------------
__global__ void absstats_kernel(const float* __restrict__ A, float* __restrict__ out2) {
    __shared__ float s_sum[256];
    __shared__ float s_sq[256];
    int tid = threadIdx.x;
    float sum = 0.f, sq = 0.f;
    for (int i = tid; i < N_CONCS * N_WL; i += 256) {
        float v = A[i];
        sum += v;
        sq  += v * v;
    }
    s_sum[tid] = sum; s_sq[tid] = sq;
    __syncthreads();
    for (int s = 128; s > 0; s >>= 1) {
        if (tid < s) { s_sum[tid] += s_sum[tid + s]; s_sq[tid] += s_sq[tid + s]; }
        __syncthreads();
    }
    if (tid == 0) {
        const float invN = 1.0f / (float)(N_CONCS * N_WL);
        float mean = s_sum[0] * invN;
        float var  = s_sq[0] * invN - mean * mean;
        out2[0] = mean;
        out2[1] = sqrtf(fmaxf(var, 0.0f));
    }
}

// ---------------------------------------------------------------------------
// Kernel 2: pack W2 [128x256] and W3 [256x128] (row-major fp32) into fp16
// WMMA B-operand layout. For a 32x16 f16 B tile: lane L -> (n = L&15,
// h = L>>4), vector element e -> k_local = h*16 + e. Tile id = n_tile*KT + k_tile,
// each tile is 32 lanes * 16 halfs = 512 halfs.
// ---------------------------------------------------------------------------
__global__ void packw_kernel(const float* __restrict__ w2, const float* __restrict__ w3,
                             _Float16* __restrict__ p2, _Float16* __restrict__ p3) {
    int t = blockIdx.x * blockDim.x + threadIdx.x;   // 0 .. 32767
    int e    = t & 15;
    int lane = (t >> 4) & 31;
    int tile = t >> 9;
    int n_in_tile = lane & 15;
    int k_local   = (lane >> 4) * 16 + e;

    // W2: 4 k_tiles x 16 n_tiles  (K=128, N=256)
    {
        int k_tile = tile & 3;
        int n_tile = tile >> 2;
        int n = n_tile * 16 + n_in_tile;
        int k = k_tile * 32 + k_local;
        p2[t] = (_Float16)w2[k * 256 + n];
    }
    // W3: 8 k_tiles x 8 n_tiles  (K=256, N=128)
    {
        int k_tile = tile & 7;
        int n_tile = tile >> 3;
        int n = n_tile * 16 + n_in_tile;
        int k = k_tile * 32 + k_local;
        p3[t] = (_Float16)w3[k * 128 + n];
    }
}

// ---------------------------------------------------------------------------
// A-operand packed-LDS address for f16 16x32 A tile:
//   k_local = 16*h + e  (h=k>>4, e=k&15)
//   lane = m + 16*h.  Index (halfs) = lane*16 + e.
// ---------------------------------------------------------------------------
__device__ __forceinline__ int a_pack_idx(int m, int k_local) {
    int h = k_local >> 4;
    int r = k_local & 15;
    return (m + 16 * h) * 16 + r;
}

// ---------------------------------------------------------------------------
// Kernel 3: main fused kernel. Each wave handles 16 batch rows.
// ---------------------------------------------------------------------------
__global__ void __launch_bounds__(WAVES * 32)
mlp_lookup_kernel(const float* __restrict__ c_norm, const float* __restrict__ wl_norm,
                  const float* __restrict__ absm,
                  const float* __restrict__ w1, const float* __restrict__ b1,
                  const float* __restrict__ b2, const float* __restrict__ b3,
                  const float* __restrict__ w4, const float* __restrict__ b4,
                  const _Float16* __restrict__ p2, const _Float16* __restrict__ p3,
                  const float* __restrict__ stats,
                  float* __restrict__ out) {
    // Per-wave activation staging, A-operand packed layout (halfs):
    __shared__ __align__(32) _Float16 bufA[WAVES][4 * 512];  // h1: 16 x 128
    __shared__ __align__(32) _Float16 bufB[WAVES][8 * 512];  // h2: 16 x 256
    __shared__ float bufC[WAVES][16 * 128];                  // h3 plain [m][k]

    const int wave = threadIdx.x >> 5;
    const int tl   = threadIdx.x & 31;
    const int tile = blockIdx.x * WAVES + wave;
    const int m0   = tile * 16;

    // ---- exact-grid lookup path (lanes 0..15, one batch row each) ----
    bool  hit = false;
    float exactv = 0.0f;
    if (tl < 16) {
        float cn = c_norm[m0 + tl];
        float wn = wl_norm[m0 + tl];
        float c  = cn * 30.0f + 30.0f;
        float wl = wn * 300.0f + 500.0f;
        int ci = (int)rintf(c * 0.25f);
        int wi = (int)rintf(wl - 200.0f);
        bool hc = (ci >= 0) && (ci < N_CONCS) && (fabsf(c - 4.0f * (float)ci) < 0.1f);
        bool hw = (wi >= 0) && (wi < N_WL)    && (fabsf(wl - 200.0f - (float)wi) < 0.1f);
        hit = hc && hw;
        int cii = hit ? ci : 0;
        int wii = hit ? wi : 0;
        exactv = (absm[cii * N_WL + wii] - stats[0]) / stats[1];
    }

    // ---- layer 1 (VALU): h1 = tanh(x @ W1 + b1), write A-packed fp16 ----
    {
        int r  = tl >> 1;            // batch row within tile
        int cb = (tl & 1) * 64;      // 64 cols per lane
        float x0 = c_norm[m0 + r];
        float x1 = wl_norm[m0 + r];
        #pragma unroll 8
        for (int j0 = 0; j0 < 64; ++j0) {
            int j = cb + j0;
            float h = fast_tanh(fmaf(x0, w1[j], fmaf(x1, w1[128 + j], b1[j])));
            int kt = j >> 5;
            bufA[wave][kt * 512 + a_pack_idx(r, j & 31)] = (_Float16)h;
        }
    }
    __syncthreads();

    const int nsel = tl & 15;   // D-matrix column handled by this lane
    const int hsel = tl >> 4;   // row-half select

    // ---- layer 2 (WMMA): h2 = tanh(h1 @ W2 + b2), K=128, N=256 ----
    {
        v16h a2[4];
        #pragma unroll
        for (int kt = 0; kt < 4; ++kt)
            a2[kt] = *(const v16h*)&bufA[wave][kt * 512 + tl * 16];

        #pragma unroll 2
        for (int nt = 0; nt < 16; ++nt) {
            float bias = b2[nt * 16 + nsel];
            v8f acc;
            #pragma unroll
            for (int i = 0; i < 8; ++i) acc[i] = bias;
            #pragma unroll
            for (int kt = 0; kt < 4; ++kt) {
                v16h bb = *(const v16h*)&p2[(nt * 4 + kt) * 512 + tl * 16];
                acc = __builtin_amdgcn_wmma_f32_16x16x32_f16(
                        false, a2[kt], false, bb, (short)0, acc, false, false);
            }
            int k = nt * 16 + nsel;               // hidden index produced
            int kt3 = k >> 5;
            #pragma unroll
            for (int reg = 0; reg < 8; ++reg) {
                int m = reg + 8 * hsel;
                bufB[wave][kt3 * 512 + a_pack_idx(m, k & 31)] = (_Float16)fast_tanh(acc[reg]);
            }
        }
    }
    __syncthreads();

    // ---- layer 3 (WMMA): h3 = tanh(h2 @ W3 + b3), K=256, N=128 ----
    {
        v16h a3[8];
        #pragma unroll
        for (int kt = 0; kt < 8; ++kt)
            a3[kt] = *(const v16h*)&bufB[wave][kt * 512 + tl * 16];

        #pragma unroll 2
        for (int nt = 0; nt < 8; ++nt) {
            float bias = b3[nt * 16 + nsel];
            v8f acc;
            #pragma unroll
            for (int i = 0; i < 8; ++i) acc[i] = bias;
            #pragma unroll
            for (int kt = 0; kt < 8; ++kt) {
                v16h bb = *(const v16h*)&p3[(nt * 8 + kt) * 512 + tl * 16];
                acc = __builtin_amdgcn_wmma_f32_16x16x32_f16(
                        false, a3[kt], false, bb, (short)0, acc, false, false);
            }
            int col = nt * 16 + nsel;
            #pragma unroll
            for (int reg = 0; reg < 8; ++reg) {
                int m = reg + 8 * hsel;
                bufC[wave][m * 128 + col] = fast_tanh(acc[reg]);
            }
        }
    }
    __syncthreads();

    // ---- layer 4 (VALU dot) + select ----
    if (tl < 16) {
        float s = b4[0];
        #pragma unroll 8
        for (int k = 0; k < 128; ++k)
            s = fmaf(bufC[wave][tl * 128 + k], w4[k], s);
        out[m0 + tl] = hit ? exactv : s;
    }
}

// ---------------------------------------------------------------------------
// Host launch
// ---------------------------------------------------------------------------
extern "C" void kernel_launch(void* const* d_in, const int* in_sizes, int n_in,
                              void* d_out, int out_size, void* d_ws, size_t ws_size,
                              hipStream_t stream) {
    const float* c_norm = (const float*)d_in[0];
    const float* wl_nrm = (const float*)d_in[1];
    // d_in[2], d_in[3]: train grids (analytic; unused)
    const float* absm   = (const float*)d_in[4];
    const float* w1     = (const float*)d_in[5];
    const float* b1     = (const float*)d_in[6];
    const float* w2     = (const float*)d_in[7];
    const float* b2     = (const float*)d_in[8];
    const float* w3     = (const float*)d_in[9];
    const float* b3     = (const float*)d_in[10];
    const float* w4     = (const float*)d_in[11];
    const float* b4     = (const float*)d_in[12];
    float*       outp   = (float*)d_out;

    float*    stats = (float*)d_ws;                                   // 2 floats
    _Float16* p2    = (_Float16*)((char*)d_ws + 256);                 // 64 KB
    _Float16* p3    = (_Float16*)((char*)d_ws + 256 + 65536);         // 64 KB

    const int B = out_size;                 // 262144

    absstats_kernel<<<1, 256, 0, stream>>>(absm, stats);
    packw_kernel<<<128, 256, 0, stream>>>(w2, w3, p2, p3);

    int tiles  = B / 16;                    // 16384 row-tiles
    int blocks = tiles / WAVES;             // 2048
    mlp_lookup_kernel<<<blocks, WAVES * 32, 0, stream>>>(
        c_norm, wl_nrm, absm, w1, b1, b2, b3, w4, b4, p2, p3, stats, outp);
}
